// WindowGCN_8031588843742
// MI455X (gfx1250) — compile-verified, matrix operand order checked by reading, not run on hardware
//
#include <hip/hip_runtime.h>
#include <hip/hip_bf16.h>

// ---------------------------------------------------------------------------
// GCN forward on gfx1250: bf16 WMMA GEMMs + L2-resident f32 atomic aggregation
// ---------------------------------------------------------------------------

#define N_NODES   100000
#define N_EDGES   1600000
#define IN_DIM    128
#define HID_DIM   256
#define OUT_DIM   256
#define N_CLASSES 200
#define N_GRAPHS  64

typedef __bf16 v8bf  __attribute__((ext_vector_type(8)));
typedef __bf16 v16bf __attribute__((ext_vector_type(16)));
typedef float  v8f   __attribute__((ext_vector_type(8)));

__device__ __forceinline__ v16bf cat8(v8bf lo, v8bf hi) {
    return __builtin_shufflevector(lo, hi, 0,1,2,3,4,5,6,7,8,9,10,11,12,13,14,15);
}

__device__ __forceinline__ __bf16 f2bf(float f) {
    union { float f; unsigned u; } v; v.f = f;
    unsigned r = v.u + 0x7FFFu + ((v.u >> 16) & 1u);   // round to nearest even
    unsigned short h = (unsigned short)(r >> 16);
    union { unsigned short s; __bf16 b; } o; o.s = h;
    return o.b;
}

// ---------------- small utility kernels ----------------

__global__ void k_fill(float* p, float val, int n) {
    int i = blockIdx.x * blockDim.x + threadIdx.x;
    if (i < n) p[i] = val;
}

__global__ void k_deg_edges(const int* __restrict__ dst, float* __restrict__ deg, int E) {
    int e = blockIdx.x * blockDim.x + threadIdx.x;
    if (e < E) atomicAdd(&deg[dst[e]], 1.0f);
}

__global__ void k_rsqrt_inplace(float* p, int n) {
    int i = blockIdx.x * blockDim.x + threadIdx.x;
    if (i < n) p[i] = rsqrtf(p[i]);          // deg >= 1 always (self loop)
}

__global__ void k_f2bf(const float* __restrict__ in, __bf16* __restrict__ out, int n) {
    int i = blockIdx.x * blockDim.x + threadIdx.x;
    if (i < n) out[i] = f2bf(in[i]);
}

// Pack row-major W[K x M] (f32) into per-lane WMMA B-fragment layout (bf16).
// Packed element index: ((kt*(M/16)+nt)*32 + lane)*16 + i
//   lane L: col n = nt*16 + (L&15), half = L>>4
//   i<8 : k = kt*32 +      half*8 + i
//   i>=8: k = kt*32 + 16 + half*8 + (i-8)
__global__ void k_pack_w(const float* __restrict__ W, __bf16* __restrict__ out, int K, int M) {
    int idx = blockIdx.x * blockDim.x + threadIdx.x;
    if (idx >= K * M) return;
    int i    = idx & 15;
    int L    = (idx >> 4) & 31;
    int tile = idx >> 9;
    int ntn  = M >> 4;
    int kt   = tile / ntn;
    int nt   = tile % ntn;
    int n    = nt * 16 + (L & 15);
    int half = L >> 4;
    int k    = (i < 8) ? (kt * 32 + half * 8 + i)
                       : (kt * 32 + 16 + half * 8 + (i - 8));
    out[idx] = f2bf(W[(size_t)k * M + n]);
}

// ---------------- WMMA GEMM: C[Nrows x 256] = A[Nrows x K] @ Bpacked ----------------
// block = 128 threads = 4 waves; wave w owns 64 output columns (4 x 16-col tiles);
// block owns 16 rows. 100000 rows = 6250 exact tiles -> EXEC all-ones (WMMA req.)
__global__ void k_gemm_bf16(const __bf16* __restrict__ A, const __bf16* __restrict__ Bp,
                            float* __restrict__ C, int K) {
    const int M    = 256;
    const int wave = threadIdx.x >> 5;
    const int lane = threadIdx.x & 31;
    const int half = lane >> 4;
    const int mr   = lane & 15;
    const int row0 = blockIdx.x * 16;

    v8f acc0 = {}, acc1 = {}, acc2 = {}, acc3 = {};
    const __bf16* Arow = A + (size_t)(row0 + mr) * K;
    const int ktiles = K >> 5;
    const int ntbase = wave * 4;                  // first 16-col tile index

    for (int kt = 0; kt < ktiles; ++kt) {
        int kb = kt * 32 + half * 8;
        v8bf alo = *(const v8bf*)(Arow + kb);
        v8bf ahi = *(const v8bf*)(Arow + kb + 16);
        v16bf a = cat8(alo, ahi);

        const __bf16* bbase = Bp + ((size_t)(kt * 16 + ntbase) * 32 + lane) * 16;
        {
            v16bf b = cat8(*(const v8bf*)(bbase + 0*512), *(const v8bf*)(bbase + 0*512 + 8));
            acc0 = __builtin_amdgcn_wmma_f32_16x16x32_bf16(false, a, false, b, (short)0, acc0, false, false);
        }
        {
            v16bf b = cat8(*(const v8bf*)(bbase + 1*512), *(const v8bf*)(bbase + 1*512 + 8));
            acc1 = __builtin_amdgcn_wmma_f32_16x16x32_bf16(false, a, false, b, (short)0, acc1, false, false);
        }
        {
            v16bf b = cat8(*(const v8bf*)(bbase + 2*512), *(const v8bf*)(bbase + 2*512 + 8));
            acc2 = __builtin_amdgcn_wmma_f32_16x16x32_bf16(false, a, false, b, (short)0, acc2, false, false);
        }
        {
            v16bf b = cat8(*(const v8bf*)(bbase + 3*512), *(const v8bf*)(bbase + 3*512 + 8));
            acc3 = __builtin_amdgcn_wmma_f32_16x16x32_bf16(false, a, false, b, (short)0, acc3, false, false);
        }
    }

    // D layout: lane L -> col n=L%16; VGPR j -> row j + 8*(L>>4)
    const int ncol = lane & 15;
    float* crow = C + (size_t)(row0 + 8 * half) * M + ntbase * 16 + ncol;
#pragma unroll
    for (int j = 0; j < 8; ++j) {
        crow[(size_t)j * M + 0]  = acc0[j];
        crow[(size_t)j * M + 16] = acc1[j];
        crow[(size_t)j * M + 32] = acc2[j];
        crow[(size_t)j * M + 48] = acc3[j];
    }
}

// ---------------- aggregation ----------------

// agg[i,f] = xw[i,f]*dinv[i]^2 + bias[f]   (self-loop contribution + bias)
__global__ void k_selfloop_bias(const float* __restrict__ xw, const float* __restrict__ dinv,
                                const float* __restrict__ bias, float* __restrict__ agg) {
    int idx = blockIdx.x * blockDim.x + threadIdx.x;   // over N_NODES*256
    int i = idx >> 8, f = idx & 255;
    float di = dinv[i];
    agg[idx] = xw[idx] * di * di + bias[f];
}

// scatter-add messages: agg[dst] += xw[src] * dinv[src]*dinv[dst]
// block = 256 threads (one per feature), 32 edges per block. E % 32 == 0.
__global__ void k_edge_agg(const int* __restrict__ src, const int* __restrict__ dst,
                           const float* __restrict__ dinv, const float* __restrict__ xw,
                           float* __restrict__ agg) {
    int e0 = blockIdx.x * 32;
    int f  = threadIdx.x;
#pragma unroll 1
    for (int k = 0; k < 32; ++k) {
        int e = e0 + k;
        int s = src[e];
        int d = dst[e];
        float nrm = dinv[s] * dinv[d];
        float v = xw[(size_t)s * 256 + f] * nrm;
        atomicAdd(agg + (size_t)d * 256 + f, v);
    }
}

__global__ void k_relu_tobf(const float* __restrict__ in, __bf16* __restrict__ out, int n) {
    int i = blockIdx.x * blockDim.x + threadIdx.x;
    if (i < n) out[i] = f2bf(fmaxf(in[i], 0.0f));
}

// ---------------- pooling + classifier ----------------

__global__ void k_pool(const float* __restrict__ agg, const int* __restrict__ batch,
                       float* __restrict__ pool) {
    int idx = blockIdx.x * blockDim.x + threadIdx.x;   // over N_NODES*256
    int i = idx >> 8, f = idx & 255;
    float v = fmaxf(agg[idx], 0.0f);                   // ReLU fused
    atomicAdd(&pool[(size_t)batch[i] * 256 + f], v);
}

__global__ void k_count(const int* __restrict__ batch, float* __restrict__ cnt, int n) {
    int i = blockIdx.x * blockDim.x + threadIdx.x;
    if (i < n) atomicAdd(&cnt[batch[i]], 1.0f);
}

__global__ void k_classifier(const float* __restrict__ pool, const float* __restrict__ cnt,
                             const float* __restrict__ Wc, const float* __restrict__ bc,
                             float* __restrict__ out) {
    int tid = blockIdx.x * blockDim.x + threadIdx.x;   // 64*200
    if (tid >= N_GRAPHS * N_CLASSES) return;
    int g = tid / N_CLASSES;
    int c = tid % N_CLASSES;
    float ic = 1.0f / fmaxf(cnt[g], 1.0f);
    float acc = bc[c];
    const float* prow = pool + (size_t)g * 256;
#pragma unroll 4
    for (int k = 0; k < OUT_DIM; ++k)
        acc += prow[k] * ic * Wc[(size_t)k * N_CLASSES + c];
    out[tid] = acc;
}

// ---------------------------------------------------------------------------

extern "C" void kernel_launch(void* const* d_in, const int* in_sizes, int n_in,
                              void* d_out, int out_size, void* d_ws, size_t ws_size,
                              hipStream_t stream) {
    const float* x    = (const float*)d_in[0];
    const int*   ei   = (const int*)d_in[1];
    const int*   batch= (const int*)d_in[2];
    const float* W1   = (const float*)d_in[4];
    const float* b1   = (const float*)d_in[5];
    const float* W2   = (const float*)d_in[6];
    const float* b2   = (const float*)d_in[7];
    const float* Wc   = (const float*)d_in[8];
    const float* bc   = (const float*)d_in[9];
    float* out = (float*)d_out;

    const int* srcp = ei;
    const int* dstp = ei + N_EDGES;

    const size_t MB = 1024ull * 1024ull;
    char* ws = (char*)d_ws;
    float*  dinv = (float*)(ws);                 // 100k f32      (0.4 MB)
    __bf16* xbf  = (__bf16*)(ws +   1 * MB);     // 100k*128 bf16 (25.6 MB)
    __bf16* hbf  = (__bf16*)(ws +  28 * MB);     // 100k*256 bf16 (51.2 MB)
    float*  xw   = (float*)(ws +  80 * MB);      // 100k*256 f32  (102.4 MB)
    float*  agg  = (float*)(ws + 184 * MB);      // 100k*256 f32  (102.4 MB)
    __bf16* w1p  = (__bf16*)(ws + 288 * MB);     // 128*256 bf16
    __bf16* w2p  = (__bf16*)(ws + 289 * MB);     // 256*256 bf16
    float*  pool = (float*)(ws + 290 * MB);      // 64*256 f32 + 64 f32
    float*  cnt  = pool + N_GRAPHS * 256;

    const int T = 256;

    // --- degrees / normalization ---
    k_fill<<<(N_NODES + T - 1) / T, T, 0, stream>>>(dinv, 1.0f, N_NODES);
    k_deg_edges<<<N_EDGES / T, T, 0, stream>>>(dstp, dinv, N_EDGES);
    k_rsqrt_inplace<<<(N_NODES + T - 1) / T, T, 0, stream>>>(dinv, N_NODES);

    // --- convert inputs / pack weights into WMMA B-fragment layout ---
    k_f2bf<<<(N_NODES * IN_DIM) / T, T, 0, stream>>>(x, xbf, N_NODES * IN_DIM);
    k_pack_w<<<(IN_DIM * HID_DIM) / T, T, 0, stream>>>(W1, w1p, IN_DIM, HID_DIM);
    k_pack_w<<<(HID_DIM * OUT_DIM) / T, T, 0, stream>>>(W2, w2p, HID_DIM, OUT_DIM);

    // --- layer 1: GEMM (WMMA bf16) -> self-loop+bias -> edge scatter -> ReLU ---
    k_gemm_bf16<<<N_NODES / 16, 128, 0, stream>>>(xbf, w1p, xw, IN_DIM);
    k_selfloop_bias<<<(N_NODES * 256) / T, T, 0, stream>>>(xw, dinv, b1, agg);
    k_edge_agg<<<N_EDGES / 32, T, 0, stream>>>(srcp, dstp, dinv, xw, agg);
    k_relu_tobf<<<(N_NODES * 256) / T, T, 0, stream>>>(agg, hbf, N_NODES * 256);

    // --- layer 2 ---
    k_gemm_bf16<<<N_NODES / 16, 128, 0, stream>>>(hbf, w2p, xw, HID_DIM);
    k_selfloop_bias<<<(N_NODES * 256) / T, T, 0, stream>>>(xw, dinv, b2, agg);
    k_edge_agg<<<N_EDGES / 32, T, 0, stream>>>(srcp, dstp, dinv, xw, agg);

    // --- global mean pool (ReLU fused) + classifier ---
    k_fill<<<(N_GRAPHS * 256 + N_GRAPHS + T - 1) / T, T, 0, stream>>>(pool, 0.0f,
                                                                      N_GRAPHS * 256 + N_GRAPHS);
    k_pool<<<(N_NODES * 256) / T, T, 0, stream>>>(agg, batch, pool);
    k_count<<<(N_NODES + T - 1) / T, T, 0, stream>>>(batch, cnt, N_NODES);
    k_classifier<<<(N_GRAPHS * N_CLASSES + T - 1) / T, T, 0, stream>>>(pool, cnt, Wc, bc, out);

    (void)in_sizes; (void)n_in; (void)out_size; (void)ws_size;
}
